// NodeTransformerLayer_55628416418470
// MI455X (gfx1250) — matrix operand order repaired and checked by reading, not compiled, not run on hardware
//
#include <hip/hip_runtime.h>
#include <hip/hip_bf16.h>

// ---------------------------------------------------------------------------
// NodeTransformerLayer for MI455X (gfx1250, wave32, WMMA 16x16x32 f16 + TDM).
//
// Algebraic refactor:
//   scores_e[b,h,i,j] = U[b,i,h,:] . LN(p)[b,i,j,:] + cb[b,i,h]
//     where U[b,i,h,:] = Wke_h^T q_s[b,i,h,:],  cb = q_s . bke_h
//   sum_j prob * v_e  = Wve_h @ (sum_j prob[h,j] * LN(p)[b,i,j,:]) + bve_h
// -> pair_emb (134 MB) is streamed exactly once (HBM-bound, ~5.8 us floor).
// The pair row is staged into LDS by the Tensor Data Mover, double-buffered
// in 32-row chunks and overlapped with the LayerNorm/f16-pack compute.
// ---------------------------------------------------------------------------

#define EMB   256
#define HEADS 8
#define HD    32
#define BB    8
#define NN    128
#define LDP   264   // padded f16 row stride for the pair tile in LDS
#define CH    32    // pair rows per TDM chunk
#define NCH   (NN / CH)

typedef __attribute__((ext_vector_type(16))) _Float16 v16h;
typedef __attribute__((ext_vector_type(8)))  float    v8f;
typedef __attribute__((ext_vector_type(4)))  unsigned int u32x4;
typedef __attribute__((ext_vector_type(8)))  int      i32x8;
typedef __attribute__((ext_vector_type(4)))  int      i32x4;

#if __has_builtin(__builtin_amdgcn_tensor_load_to_lds) && \
    __has_builtin(__builtin_amdgcn_s_wait_tensorcnt)
#define USE_TDM 1
#else
#define USE_TDM 0
#endif

// ---- WMMA helpers ----------------------------------------------------------

__device__ __forceinline__ v8f wmma16(v16h a, v16h b, v8f c) {
  return __builtin_amdgcn_wmma_f32_16x16x32_f16(false, a, false, b,
                                                (short)0, c, false, false);
}

// A fragment (16x32 f16), row-major source, tile origin p, leading dim ld.
// Contiguous k order used here (B matches): lane&15 = M row; K = 2v + hi*16
__device__ __forceinline__ v16h load_a(const _Float16* p, int ld) {
  int lane = threadIdx.x & 31;
  int m = lane & 15, hi = lane >> 4;
  v16h a;
#pragma unroll
  for (int v = 0; v < 8; ++v) {
    int k = 2 * v + hi * 16;
    a[2 * v]     = p[m * ld + k];
    a[2 * v + 1] = p[m * ld + k + 1];
  }
  return a;
}

// B fragment (32x16 f16) where B[k][n] = p[n*ld + k]  (transposed source, e.g.
// weight stored row-major as W[n][k]).  Contiguous k order (matches A): lane&15 = N col; K = 2v + hi*16.
__device__ __forceinline__ v16h load_b_T(const _Float16* p, int ld) {
  int lane = threadIdx.x & 31;
  int n = lane & 15, hi = lane >> 4;
  v16h b;
#pragma unroll
  for (int v = 0; v < 8; ++v) {
    int k = 2 * v + hi * 16;
    b[2 * v]     = p[n * ld + k];
    b[2 * v + 1] = p[n * ld + k + 1];
  }
  return b;
}

// B fragment (32x16 f16) where B[k][n] = p[k*ld + n]  (direct K-major source).
__device__ __forceinline__ v16h load_b_D(const _Float16* p, int ld) {
  int lane = threadIdx.x & 31;
  int n = lane & 15, hi = lane >> 4;
  v16h b;
#pragma unroll
  for (int v = 0; v < 8; ++v) {
    int k = 2 * v + hi * 16;
    b[2 * v]     = p[k * ld + n];
    b[2 * v + 1] = p[(k + 1) * ld + n];
  }
  return b;
}

// ---- TDM: issue a 2-D tensor tile load (CH x 256 f32, contiguous) ----------

#if USE_TDM
__device__ __forceinline__ void tdm_issue_chunk(const float* gsrc,
                                                unsigned lds_byte_off) {
  size_t ga = (size_t)gsrc;
  u32x4 g0;
  g0[0] = 1u;                                  // count=1, user descriptor
  g0[1] = lds_byte_off;                        // D#.lds_addr
  g0[2] = (unsigned)(ga & 0xffffffffu);        // global_addr[31:0]
  g0[3] = (unsigned)(ga >> 32) | 0x80000000u;  // global_addr[56:32] | type=2
  i32x8 g1;
  g1[0] = 0x20000;                 // workgroup_mask=0, data_size=2 (4 bytes)
  g1[1] = (int)(256u << 16);       // tensor_dim0 = 256  (bits 79:48, low half)
  g1[2] = (int)(CH << 16);         // dim0 hi = 0 | tensor_dim1 = CH
  g1[3] = (int)(256u << 16);       // dim1 hi = 0 | tile_dim0 = 256
  g1[4] = CH;                      // tile_dim1 = CH, tile_dim2 = 0
  g1[5] = 256;                     // tensor_dim0_stride = 256 elements
  g1[6] = 0;
  g1[7] = 0;
  i32x4 z4 = {0, 0, 0, 0};
  i32x8 z8 = {0, 0, 0, 0, 0, 0, 0, 0};
  // clang-23 / therock-10.0 lane: 6-arg form (extra i32x8 group, then cpol)
  __builtin_amdgcn_tensor_load_to_lds(g0, g1, z4, z4, z8, 0);
}
#endif

// ---- Kernel 0: f32 -> f16 weight staging (Wq|Wk|Wv|Wke contiguous) ---------

__global__ __launch_bounds__(256) void cvt_weights_kernel(
    const float* __restrict__ wq, const float* __restrict__ wk,
    const float* __restrict__ wv, const float* __restrict__ wke,
    _Float16* __restrict__ dst) {
  int idx = blockIdx.x * 256 + threadIdx.x;      // 0..65535
  dst[idx]              = (_Float16)wq[idx];
  dst[65536 + idx]      = (_Float16)wk[idx];
  dst[2 * 65536 + idx]  = (_Float16)wv[idx];
  dst[3 * 65536 + idx]  = (_Float16)wke[idx];
}

// ---- Kernel 1: LN(atom) + q/k/v projections + U + cb -----------------------
// One block = 16 node-rows (flattened over B*N = 1024). 256 threads = 8 waves.

__global__ __launch_bounds__(256) void proj_kernel(
    const float* __restrict__ atom,
    const float* __restrict__ gn, const float* __restrict__ bn,
    const float* __restrict__ bq, const float* __restrict__ bk,
    const float* __restrict__ bv, const float* __restrict__ bke,
    const _Float16* __restrict__ W16,
    float* __restrict__ v_ws,
    _Float16* __restrict__ q16_ws, _Float16* __restrict__ k16_ws,
    _Float16* __restrict__ U16_ws, float* __restrict__ cb_ws) {
  __shared__ _Float16 x16[16 * 256];
  __shared__ _Float16 q16[16 * 256];
  __shared__ float    qf[16 * 256];
  __shared__ float    lnm[16], lnr[16];

  int t = threadIdx.x;
  int mbase = blockIdx.x * 16;

  if (t < 16) {
    const float* r = atom + (size_t)(mbase + t) * EMB;
    float s = 0.f, s2 = 0.f;
    for (int e = 0; e < EMB; ++e) { float a = r[e]; s += a; s2 += a * a; }
    float m = s * (1.f / EMB);
    float var = s2 * (1.f / EMB) - m * m;
    lnm[t] = m;
    lnr[t] = rsqrtf(var + 1e-5f);
  }
  __syncthreads();
  for (int idx = t; idx < 16 * EMB; idx += 256) {
    int row = idx >> 8, e = idx & 255;
    float a = atom[(size_t)(mbase + row) * EMB + e];
    x16[idx] = (_Float16)((a - lnm[row]) * lnr[row] * gn[e] + bn[e]);
  }
  __syncthreads();

  int w = t >> 5;
  int lane = t & 31, n = lane & 15, hi = lane >> 4;
  const float scale = 0.1767766952966369f; // 1/sqrt(32)

  for (int proj = 0; proj < 3; ++proj) {
    const _Float16* W = W16 + proj * 65536;
    for (int ti = 0; ti < 2; ++ti) {
      int nt = w + ti * 8;                       // 16 n-tiles over 8 waves
      v8f acc = {};
      for (int kt = 0; kt < 8; ++kt) {
        v16h a  = load_a(x16 + kt * 32, 256);
        v16h bb = load_b_T(W + (nt * 16) * 256 + kt * 32, 256);
        acc = wmma16(a, bb, acc);
      }
      int j = nt * 16 + n;
#pragma unroll
      for (int r = 0; r < 8; ++r) {
        int m = r + hi * 8;
        size_t go = (size_t)(mbase + m) * EMB + j;
        float val = acc[r];
        if (proj == 0) {
          val = (val + bq[j]) * scale;
          qf[m * 256 + j] = val;
          _Float16 h16 = (_Float16)val;
          q16[m * 256 + j] = h16;
          q16_ws[go] = h16;
        } else if (proj == 1) {
          k16_ws[go] = (_Float16)(val + bk[j]);
        } else {
          v_ws[go] = val + bv[j];
        }
      }
    }
  }
  __syncthreads();

  // U[m, h, :] = q_s[m, h*32..] @ Wke_h   (head h = wave id, K = 32, one step)
  {
    int h = w;
    v16h a = load_a(q16 + h * 32, 256);
    for (int et = 0; et < 16; ++et) {
      v16h bb = load_b_D(W16 + 3 * 65536 + (h * 32) * 256 + et * 16, 256);
      v8f acc = {};
      acc = wmma16(a, bb, acc);
#pragma unroll
      for (int r = 0; r < 8; ++r) {
        int m = r + hi * 8;
        U16_ws[(size_t)(mbase + m) * (HEADS * EMB) + h * EMB + et * 16 + n] =
            (_Float16)acc[r];
      }
    }
  }
  if (t < 128) {
    int m = t >> 3, h = t & 7;
    float s = 0.f;
    for (int d = 0; d < HD; ++d) s += qf[m * 256 + h * 32 + d] * bke[h * 32 + d];
    cb_ws[(mbase + m) * HEADS + h] = s;
  }
}

// ---- Kernel 1b: scores_n[b,h,i,j] = q_i . k_j  (pure WMMA, K = 32) ---------

__global__ __launch_bounds__(32) void scores_n_kernel(
    const _Float16* __restrict__ q16_ws, const _Float16* __restrict__ k16_ws,
    float* __restrict__ sn_ws) {
  int bid = blockIdx.x;        // 0..511
  int b  = bid >> 6;
  int h  = (bid >> 3) & 7;
  int it = bid & 7;
  int lane = threadIdx.x & 31, n = lane & 15, hi = lane >> 4;

  v16h a = load_a(q16_ws + (size_t)(b * NN + it * 16) * EMB + h * 32, EMB);
  for (int jt = 0; jt < 8; ++jt) {
    v16h bb = load_b_T(k16_ws + (size_t)(b * NN + jt * 16) * EMB + h * 32, EMB);
    v8f acc = {};
    acc = wmma16(a, bb, acc);
#pragma unroll
    for (int r = 0; r < 8; ++r) {
      int m = r + hi * 8;
      sn_ws[((size_t)(b * HEADS + h) * NN + it * 16 + m) * NN + jt * 16 + n] =
          acc[r];
    }
  }
}

// ---- Kernel 2: TDM-stream one pair row, LN + scores_e + softmax + Pw + out -
// grid = B*N = 1024 blocks, 256 threads (8 waves), ~153 KB dynamic LDS.

__global__ __launch_bounds__(256) void pair_attn_kernel(
    const float* __restrict__ pair_emb, const float* __restrict__ pair_mask,
    const float* __restrict__ node_mask,
    const float* __restrict__ gpg, const float* __restrict__ bpg,
    const float* __restrict__ Wve, const float* __restrict__ bve,
    const float* __restrict__ Wva, const float* __restrict__ bva,
    const _Float16* __restrict__ U16_ws, const float* __restrict__ cb_ws,
    const float* __restrict__ sn_ws, const float* __restrict__ v_ws,
    float* __restrict__ out) {
  extern __shared__ char smem_raw[];
  _Float16* pn   = (_Float16*)smem_raw;           // [128][LDP] normalized pair
  _Float16* u16  = pn + NN * LDP;                 // [16][256], rows 8..15 = 0
  float*    raw0 = (float*)(u16 + 16 * 256);      // [CH][256] raw f32 chunk
  float*    raw1 = raw0 + CH * EMB;               // double buffer
  float*    sc   = raw1 + CH * EMB;               // [8][128] scores -> probs
  float*    Pw   = sc + HEADS * NN;               // [8][256]
  float*    mrow = Pw + HEADS * EMB;              // [128]
  float*    rrow = mrow + NN;                     // [128]
  float*    gp   = rrow + NN;                     // [256]
  float*    bp   = gp + EMB;                      // [256]
  float*    combF = gp;                           // reuse at epilogue

  int t = threadIdx.x;
  int lane = t & 31;
  int w = t >> 5;
  int bid = blockIdx.x;
  int b = bid >> 7;
  int i = bid & 127;
  size_t rowbase = (size_t)(b * NN + i);

  gp[t] = gpg[t];
  bp[t] = bpg[t];

  const float* prow = pair_emb + rowbase * NN * EMB;
  const _Float16* Ubase = U16_ws + rowbase * (HEADS * EMB);
  for (int idx = t; idx < 16 * EMB; idx += 256)
    u16[idx] = (idx < HEADS * EMB) ? Ubase[idx] : (_Float16)0.f;

#if USE_TDM
  unsigned raw_off[2];
  raw_off[0] = (unsigned)(size_t)raw0;   // LDS aperture: low 32 bits = offset
  raw_off[1] = (unsigned)(size_t)raw1;
  if (t < 32) tdm_issue_chunk(prow, raw_off[0]);          // prefetch chunk 0
#endif

  // ---- pipelined LN over NCH chunks of CH rows ----
  for (int c = 0; c < NCH; ++c) {
    float* raw = (c & 1) ? raw1 : raw0;
#if USE_TDM
    if (t < 32) {
      if (c + 1 < NCH) {
        tdm_issue_chunk(prow + (size_t)(c + 1) * CH * EMB, raw_off[(c + 1) & 1]);
        __builtin_amdgcn_s_wait_tensorcnt(1);   // chunk c landed (in-order)
      } else {
        __builtin_amdgcn_s_wait_tensorcnt(0);
      }
    }
#else
    for (int idx = t; idx < CH * EMB; idx += 256)
      raw[idx] = prow[(size_t)c * CH * EMB + idx];
#endif
    __syncthreads();

    // stats: 8 lanes per row, shfl-reduce within the 8-lane group
    {
      int r = t >> 3, p = t & 7;                // r in 0..31
      const float* rr = raw + r * EMB + p * 32;
      float s = 0.f, s2 = 0.f;
#pragma unroll
      for (int e = 0; e < 32; ++e) { float a = rr[e]; s += a; s2 += a * a; }
#pragma unroll
      for (int off = 1; off < 8; off <<= 1) {
        s  += __shfl_xor(s,  off, 32);
        s2 += __shfl_xor(s2, off, 32);
      }
      if (p == 0) {
        float m = s * (1.f / EMB);
        float var = s2 * (1.f / EMB) - m * m;
        mrow[c * CH + r] = m;
        rrow[c * CH + r] = rsqrtf(var + 1e-5f);
      }
    }
    __syncthreads();

    // normalize chunk (f32 in LDS) -> pn f16
    for (int idx = t; idx < CH * EMB; idx += 256) {
      int j = idx >> 8, e = idx & 255;
      int jj = c * CH + j;
      float f = (raw[idx] - mrow[jj]) * rrow[jj] * gp[e] + bp[e];
      pn[jj * LDP + e] = (_Float16)f;
    }
    __syncthreads();   // raw buffer reusable by the issue in the next iter
  }

  // init scores with scores_n + cb + mask
  for (int idx = t; idx < HEADS * NN; idx += 256) {
    int h = idx >> 7, j = idx & 127;
    float msk = pair_mask[rowbase * NN + j];
    sc[idx] = sn_ws[((size_t)(b * HEADS + h) * NN + i) * NN + j] +
              cb_ws[rowbase * HEADS + h] + (1.f - msk) * -1e6f;
  }
  __syncthreads();

  // scores_e: Pn[128,256] @ U^T[256,16(8 used)] -> add into sc. wave w = j-tile.
  {
    int jt = w;
    v8f acc = {};
    for (int kt = 0; kt < 8; ++kt) {
      v16h a  = load_a(pn + jt * 16 * LDP + kt * 32, LDP);
      v16h bb = load_b_T(u16 + kt * 32, 256);
      acc = wmma16(a, bb, acc);
    }
    int n = lane & 15, hi = lane >> 4;
    if (n < HEADS) {
#pragma unroll
      for (int r = 0; r < 8; ++r) {
        int m = r + hi * 8;
        sc[n * NN + jt * 16 + m] += acc[r];
      }
    }
  }
  __syncthreads();

  // softmax: wave w owns head h = w
  {
    int h = w;
    float vals[4], mx = -3.4e38f;
#pragma unroll
    for (int c = 0; c < 4; ++c) {
      vals[c] = sc[h * NN + lane + c * 32];
      mx = fmaxf(mx, vals[c]);
    }
#pragma unroll
    for (int off = 16; off > 0; off >>= 1) mx = fmaxf(mx, __shfl_xor(mx, off, 32));
    float sum = 0.f;
#pragma unroll
    for (int c = 0; c < 4; ++c) { vals[c] = __expf(vals[c] - mx); sum += vals[c]; }
#pragma unroll
    for (int off = 16; off > 0; off >>= 1) sum += __shfl_xor(sum, off, 32);
    float inv = 1.f / sum;
#pragma unroll
    for (int c = 0; c < 4; ++c) sc[h * NN + lane + c * 32] = vals[c] * inv;
  }
  __syncthreads();

  // Pw[8,256] = prob[8,128] @ Pn[128,256]  (A rows 8..15 zero-padded)
  {
    int m16 = lane & 15, hi = lane >> 4;
    for (int ei = 0; ei < 2; ++ei) {
      int et = w * 2 + ei;                       // 16 e-tiles over 8 waves
      v8f acc = {};
      for (int kt = 0; kt < 4; ++kt) {
        v16h a;
#pragma unroll
        for (int v = 0; v < 8; ++v) {
          int k = 2 * v + hi * 16;
          float p0 = (m16 < HEADS) ? sc[m16 * NN + kt * 32 + k] : 0.f;
          float p1 = (m16 < HEADS) ? sc[m16 * NN + kt * 32 + k + 1] : 0.f;
          a[2 * v]     = (_Float16)p0;
          a[2 * v + 1] = (_Float16)p1;
        }
        v16h bb = load_b_D(pn + (kt * 32) * LDP + et * 16, LDP);
        acc = wmma16(a, bb, acc);
      }
#pragma unroll
      for (int r = 0; r < 8; ++r) {
        int m = r + hi * 8;
        if (m < HEADS) Pw[m * EMB + et * 16 + (lane & 15)] = acc[r];
      }
    }
  }
  __syncthreads();

  // combined[h,d] = (sum_j prob*v + v_i) + (Wve_h @ Pw_h + bve)   (s == 1)
  float comb;
  {
    int h = w;                                   // t = h*32 + d
    float accn = v_ws[rowbase * EMB + t];
    const float* sch = sc + h * NN;
    for (int j = 0; j < NN; ++j)
      accn += sch[j] * v_ws[((size_t)(b * NN + j)) * EMB + t];
    float acce = bve[t];
    const float* wver = Wve + (size_t)t * EMB;
    const float* pwh = Pw + h * EMB;
    for (int e = 0; e < EMB; ++e) acce += wver[e] * pwh[e];
    comb = accn + acce;
  }
  combF[t] = comb;
  __syncthreads();

  // out = combined @ Wva.T + bva, masked
  {
    int h = w, d = t & 31;
    float o = bva[d];
    const float* wvr = Wva + d * HD;
    const float* ch = combF + h * HD;
#pragma unroll
    for (int d2 = 0; d2 < HD; ++d2) o += ch[d2] * wvr[d2];
    o *= node_mask[rowbase];
    out[rowbase * EMB + t] = o;
  }
}

// ---------------------------------------------------------------------------

extern "C" void kernel_launch(void* const* d_in, const int* in_sizes, int n_in,
                              void* d_out, int out_size, void* d_ws, size_t ws_size,
                              hipStream_t stream) {
  (void)in_sizes; (void)n_in; (void)out_size; (void)ws_size;

  const float* atom  = (const float*)d_in[0];
  const float* pair  = (const float*)d_in[1];
  const float* nmask = (const float*)d_in[2];
  const float* pmask = (const float*)d_in[3];
  const float* gn    = (const float*)d_in[4];
  const float* bn    = (const float*)d_in[5];
  const float* gpp   = (const float*)d_in[6];
  const float* bpp   = (const float*)d_in[7];
  const float* Wq    = (const float*)d_in[8];
  const float* bq    = (const float*)d_in[9];
  const float* Wk    = (const float*)d_in[10];
  const float* bk    = (const float*)d_in[11];
  const float* Wv    = (const float*)d_in[12];
  const float* bv    = (const float*)d_in[13];
  const float* Wke   = (const float*)d_in[14];
  const float* bke   = (const float*)d_in[15];
  const float* Wve   = (const float*)d_in[16];
  const float* bve   = (const float*)d_in[17];
  const float* Wva   = (const float*)d_in[18];
  const float* bva   = (const float*)d_in[19];
  float* out = (float*)d_out;

  // workspace layout (bytes)
  char* ws = (char*)d_ws;
  _Float16* W16    = (_Float16*)(ws + 0);          //  512 KB: Wq|Wk|Wv|Wke f16
  float*    v_ws   = (float*)   (ws + 524288);     // 1 MB: v projection f32
  _Float16* q16_ws = (_Float16*)(ws + 1572864);    //  512 KB
  _Float16* k16_ws = (_Float16*)(ws + 2097152);    //  512 KB
  _Float16* U16_ws = (_Float16*)(ws + 2621440);    // 4 MB: U[b,n,h,e] f16
  float*    cb_ws  = (float*)   (ws + 6815744);    //   32 KB
  float*    sn_ws  = (float*)   (ws + 6848512);    // 4 MB: q.k^T scores

  cvt_weights_kernel<<<256, 256, 0, stream>>>(Wq, Wk, Wv, Wke, W16);
  proj_kernel<<<64, 256, 0, stream>>>(atom, gn, bn, bq, bk, bv, bke, W16,
                                      v_ws, q16_ws, k16_ws, U16_ws, cb_ws);
  scores_n_kernel<<<512, 32, 0, stream>>>(q16_ws, k16_ws, sn_ws);

  size_t shmem = (size_t)(NN * LDP + 16 * 256) * sizeof(_Float16) +
                 (size_t)(2 * CH * EMB + HEADS * NN + HEADS * EMB +
                          NN + NN + EMB + EMB) * sizeof(float);  // 156672 B
  pair_attn_kernel<<<BB * NN, 256, shmem, stream>>>(
      pair, pmask, nmask, gpp, bpp, Wve, bve, Wva, bva,
      U16_ws, cb_ws, sn_ws, v_ws, out);
}
